// DualAttentionModule_4088808866179
// MI455X (gfx1250) — compile-verified, matrix operand order checked by reading, not run on hardware
//
#include <hip/hip_runtime.h>
#include <math.h>

// ---------------------------------------------------------------------------
// DualAttention on gfx1250: flash-style two-pass attention, bf16 WMMA,
// f32 accumulate. No HW x HW matrix is ever materialized.
// ---------------------------------------------------------------------------

typedef __attribute__((ext_vector_type(16))) __bf16 bf16x16;
typedef __attribute__((ext_vector_type(8)))  __bf16 bf16x8;
typedef __attribute__((ext_vector_type(8)))  float  f32x8;

static constexpr int kC  = 128;    // channels
static constexpr int kHW = 4096;   // 64*64 spatial
static constexpr int kB  = 2;      // batch

// Load one bf16 WMMA operand fragment (A- or B-style, identical per-lane
// pattern) from a row-major (rows x rowStride) bf16 array.
// Per-lane element e holds column col0 + k(e,half), k(e,h)=e+8h (+8 if e>=8):
// two contiguous 16-byte chunks.
__device__ __forceinline__ bf16x16 load_frag(const __bf16* __restrict__ base,
                                             int row, int rowStride,
                                             int col0, int half) {
  const __bf16* p = base + (size_t)row * rowStride + col0 + half * 8;
  bf16x8 lo = *(const bf16x8*)(p);
  bf16x8 hi = *(const bf16x8*)(p + 16);
  return __builtin_shufflevector(lo, hi, 0, 1, 2, 3, 4, 5, 6, 7,
                                          8, 9, 10, 11, 12, 13, 14, 15);
}

__device__ __forceinline__ f32x8 wmma_bf16(bf16x16 a, bf16x16 b, f32x8 c) {
  // (neg_a, A, neg_b, B, c_mod, C, reuse_a, reuse_b)
  return __builtin_amdgcn_wmma_f32_16x16x32_bf16(false, a, false, b,
                                                 (short)0, c, false, false);
}

// ---------------------------------------------------------------------------
// Kernel 1: 1x1 convs f,g,h from x, plus bf16 operand layouts.
//   Ft,Gt,Xt : (HW x C) row-major bf16   (S-GEMM operands)
//   Hc,Xc    : (C x HW) row-major bf16   (P*V A-operands)
// ---------------------------------------------------------------------------
__global__ void __launch_bounds__(256)
conv_prep_kernel(const float* __restrict__ x,
                 const float* __restrict__ fw, const float* __restrict__ fb,
                 const float* __restrict__ gw, const float* __restrict__ gb,
                 const float* __restrict__ hw_, const float* __restrict__ hb,
                 __bf16* __restrict__ Ft, __bf16* __restrict__ Gt,
                 __bf16* __restrict__ Hc, __bf16* __restrict__ Xt,
                 __bf16* __restrict__ Xc) {
  const int b  = blockIdx.y;
  const int i0 = blockIdx.x * 64;
  const float* X = x + (size_t)b * kC * kHW;

  __shared__ float Xs[kC][64];
  for (int idx = threadIdx.x; idx < kC * 64; idx += 256) {
    int c = idx >> 6, ii = idx & 63;
    Xs[c][ii] = X[(size_t)c * kHW + i0 + ii];
  }
  __syncthreads();

  __bf16* FtB = Ft + (size_t)b * kHW * kC;
  __bf16* GtB = Gt + (size_t)b * kHW * kC;
  __bf16* HcB = Hc + (size_t)b * kC * kHW;
  __bf16* XtB = Xt + (size_t)b * kHW * kC;
  __bf16* XcB = Xc + (size_t)b * kC * kHW;

  for (int idx = threadIdx.x; idx < kC * 64; idx += 256) {
    int c = idx >> 6, ii = idx & 63;
    __bf16 bv = (__bf16)Xs[c][ii];
    XcB[(size_t)c * kHW + i0 + ii] = bv;
    XtB[(size_t)(i0 + ii) * kC + c] = bv;
  }

  const int ii = threadIdx.x & 63;
  const int og = threadIdx.x >> 6;          // 0..3
  for (int oo = 0; oo < 32; ++oo) {
    const int o = og + (oo << 2);
    float af = fb[o], ag = gb[o], ah = hb[o];
    #pragma unroll 8
    for (int c = 0; c < kC; ++c) {
      float xv = Xs[c][ii];
      af = fmaf(fw[o * kC + c], xv, af);
      ag = fmaf(gw[o * kC + c], xv, ag);
      ah = fmaf(hw_[o * kC + c], xv, ah);
    }
    FtB[(size_t)(i0 + ii) * kC + o] = (__bf16)af;
    GtB[(size_t)(i0 + ii) * kC + o] = (__bf16)ag;
    HcB[(size_t)o * kHW + i0 + ii]  = (__bf16)ah;
  }
}

// ---------------------------------------------------------------------------
// Kernel 2: softmax row stats (running max m and reciprocal sum 1/d).
// One wave handles 16 query columns; WMMA computes S transposed
// (rows=key j, cols=query i); each lane reduces its 8 rows per tile online,
// halves combined with a lane-xor-16 shuffle.
//   which==0: position  (Key=Gt, Qry=Ft) -> mp, rdp
//   which==1: channel   (Key=Xt, Qry=Xt) -> mc, rdc   (S symmetric)
// ---------------------------------------------------------------------------
__global__ void __launch_bounds__(256)
attn_stats_kernel(const __bf16* __restrict__ Gt, const __bf16* __restrict__ Ft,
                  const __bf16* __restrict__ Xt,
                  float* __restrict__ mp, float* __restrict__ rdp,
                  float* __restrict__ mc, float* __restrict__ rdc) {
  const int b = blockIdx.y, which = blockIdx.z;
  const __bf16* Key = (which ? Xt : Gt) + (size_t)b * kHW * kC;
  const __bf16* Qry = (which ? Xt : Ft) + (size_t)b * kHW * kC;
  float* M = (which ? mc : mp) + (size_t)b * kHW;
  float* R = (which ? rdc : rdp) + (size_t)b * kHW;

  const int wave = threadIdx.x >> 5;
  const int lane = threadIdx.x & 31;
  const int r = lane & 15, half = lane >> 4;
  const int i0 = (blockIdx.x * 8 + wave) * 16;

  bf16x16 Bq[4];
  #pragma unroll
  for (int cs = 0; cs < 4; ++cs) Bq[cs] = load_frag(Qry, i0 + r, kC, cs * 32, half);

  float m = -INFINITY, d = 0.f;
  for (int j0 = 0; j0 < kHW; j0 += 16) {
    f32x8 s = {};
    #pragma unroll
    for (int cs = 0; cs < 4; ++cs) {
      bf16x16 A = load_frag(Key, j0 + r, kC, cs * 32, half);
      s = wmma_bf16(A, Bq[cs], s);
    }
    float nm = m;
    #pragma unroll
    for (int v = 0; v < 8; ++v) nm = fmaxf(nm, s[v]);
    d *= __expf(m - nm);
    #pragma unroll
    for (int v = 0; v < 8; ++v) d += __expf(s[v] - nm);
    m = nm;
  }
  // combine lane L with lane L^16 (other half of the j range)
  float mo = __shfl_xor(m, 16, 32);
  float dd = __shfl_xor(d, 16, 32);
  float Mx = fmaxf(m, mo);
  float D  = d * __expf(m - Mx) + dd * __expf(mo - Mx);
  if (half == 0) {
    M[i0 + r] = Mx;
    R[i0 + r] = 1.0f / D;
  }
}

// ---------------------------------------------------------------------------
// Kernel 3: apply attention. One wave owns 16 output columns and a
// (128 x 16) f32 accumulator O^T (8 C-fragments). Per 32-key block:
// recompute two S^T tiles, convert element-wise (exp * 1/d -> bf16) directly
// into a B-operand fragment P (no shuffles), then 8 WMMAs of V*P.
//   which==0: position  Out = h . P^T   (scale by query/col stats)
//   which==1: channel   Out = x . P     (scale by key/row stats; S symmetric)
// ---------------------------------------------------------------------------
__global__ void __launch_bounds__(256)
attn_apply_kernel(const __bf16* __restrict__ Gt, const __bf16* __restrict__ Ft,
                  const __bf16* __restrict__ Hc, const __bf16* __restrict__ Xt,
                  const __bf16* __restrict__ Xc,
                  const float* __restrict__ mp, const float* __restrict__ rdp,
                  const float* __restrict__ mc, const float* __restrict__ rdc,
                  float* __restrict__ Otp, float* __restrict__ Otc) {
  const int b = blockIdx.y, which = blockIdx.z;
  const __bf16* Key = (which ? Xt : Gt) + (size_t)b * kHW * kC;
  const __bf16* Qry = (which ? Xt : Ft) + (size_t)b * kHW * kC;
  const __bf16* V   = (which ? Xc : Hc) + (size_t)b * kC * kHW;
  const float* M = (which ? mc : mp) + (size_t)b * kHW;
  const float* R = (which ? rdc : rdp) + (size_t)b * kHW;
  float* Out = (which ? Otc : Otp) + (size_t)b * kC * kHW;

  const int wave = threadIdx.x >> 5;
  const int lane = threadIdx.x & 31;
  const int r = lane & 15, half = lane >> 4;
  const int i0 = (blockIdx.x * 8 + wave) * 16;

  bf16x16 Bq[4];
  #pragma unroll
  for (int cs = 0; cs < 4; ++cs) Bq[cs] = load_frag(Qry, i0 + r, kC, cs * 32, half);

  const float mq = M[i0 + r];   // per-lane query stats (position path)
  const float rq = R[i0 + r];

  f32x8 acc[8];
  #pragma unroll
  for (int ct = 0; ct < 8; ++ct) acc[ct] = (f32x8){};

  for (int j0 = 0; j0 < kHW; j0 += 32) {
    const int jn = (j0 + 32 < kHW) ? (j0 + 32) : 0;
    __builtin_prefetch(Key + (size_t)(jn + r) * kC, 0, 1);
    __builtin_prefetch(V + (size_t)r * kHW + jn, 0, 1);

    bf16x16 P = {};
    #pragma unroll
    for (int t = 0; t < 2; ++t) {
      const int jt0 = j0 + t * 16;
      f32x8 s = {};
      #pragma unroll
      for (int cs = 0; cs < 4; ++cs) {
        bf16x16 A = load_frag(Key, jt0 + r, kC, cs * 32, half);
        s = wmma_bf16(A, Bq[cs], s);
      }
      if (which) {
        // channel: normalize along the key index (fragment rows)
        const float4* m4 = (const float4*)(M + jt0 + half * 8);
        const float4* r4 = (const float4*)(R + jt0 + half * 8);
        float4 ma = m4[0], mb = m4[1], ra = r4[0], rb = r4[1];
        float mv[8] = {ma.x, ma.y, ma.z, ma.w, mb.x, mb.y, mb.z, mb.w};
        float rv[8] = {ra.x, ra.y, ra.z, ra.w, rb.x, rb.y, rb.z, rb.w};
        #pragma unroll
        for (int v = 0; v < 8; ++v)
          P[t * 8 + v] = (__bf16)(__expf(s[v] - mv[v]) * rv[v]);
      } else {
        // position: normalize along the query index (per-lane scalar)
        #pragma unroll
        for (int v = 0; v < 8; ++v)
          P[t * 8 + v] = (__bf16)(__expf(s[v] - mq) * rq);
      }
    }
    #pragma unroll
    for (int ct = 0; ct < 8; ++ct) {
      bf16x16 Av = load_frag(V, ct * 16 + r, kHW, j0, half);
      acc[ct] = wmma_bf16(Av, P, acc[ct]);
    }
  }

  // Store O^T: lane column i = i0+r, rows c = ct*16 + v + 8*half
  #pragma unroll
  for (int ct = 0; ct < 8; ++ct) {
    #pragma unroll
    for (int v = 0; v < 8; ++v) {
      int c = ct * 16 + v + 8 * half;
      Out[(size_t)c * kHW + i0 + r] = acc[ct][v];
    }
  }
}

// ---------------------------------------------------------------------------
// Kernel 4: y = out_w @ concat(Otp + x, Otc + x) + out_b   (f32, LDS-staged)
// ---------------------------------------------------------------------------
__global__ void __launch_bounds__(256)
fuse_kernel(const float* __restrict__ x,
            const float* __restrict__ Otp, const float* __restrict__ Otc,
            const float* __restrict__ ow, const float* __restrict__ ob,
            float* __restrict__ y) {
  const int b = blockIdx.y;
  const int i0 = blockIdx.x * 32;
  const float* X = x + (size_t)b * kC * kHW;
  const float* P = Otp + (size_t)b * kC * kHW;
  const float* Cc = Otc + (size_t)b * kC * kHW;

  __shared__ float Ps[kC][32];
  __shared__ float Cs[kC][32];
  for (int idx = threadIdx.x; idx < kC * 32; idx += 256) {
    int c = idx >> 5, ii = idx & 31;
    float xv = X[(size_t)c * kHW + i0 + ii];
    Ps[c][ii] = P[(size_t)c * kHW + i0 + ii] + xv;
    Cs[c][ii] = Cc[(size_t)c * kHW + i0 + ii] + xv;
  }
  __syncthreads();

  const int ii = threadIdx.x & 31;
  const int og = threadIdx.x >> 5;          // 0..7
  for (int oo = 0; oo < 16; ++oo) {
    const int o = og + (oo << 3);
    float a = ob[o];
    #pragma unroll 8
    for (int c = 0; c < kC; ++c) {
      a = fmaf(ow[o * 2 * kC + c],      Ps[c][ii], a);
      a = fmaf(ow[o * 2 * kC + kC + c], Cs[c][ii], a);
    }
    y[(size_t)b * kC * kHW + (size_t)o * kHW + i0 + ii] = a;
  }
}

// ---------------------------------------------------------------------------
// Host-side launch
// ---------------------------------------------------------------------------
extern "C" void kernel_launch(void* const* d_in, const int* in_sizes, int n_in,
                              void* d_out, int out_size, void* d_ws,
                              size_t ws_size, hipStream_t stream) {
  const float* x  = (const float*)d_in[0];
  const float* fw = (const float*)d_in[1];
  const float* fb = (const float*)d_in[2];
  const float* gw = (const float*)d_in[3];
  const float* gb = (const float*)d_in[4];
  const float* hw = (const float*)d_in[5];
  const float* hb = (const float*)d_in[6];
  const float* ow = (const float*)d_in[7];
  const float* ob = (const float*)d_in[8];
  float* y = (float*)d_out;

  // workspace layout (~18.2 MB total)
  char* w = (char*)d_ws;
  const size_t bfArr = (size_t)kB * kHW * kC * sizeof(__bf16);   // 2 MB each
  __bf16* Ft = (__bf16*)w; w += bfArr;
  __bf16* Gt = (__bf16*)w; w += bfArr;
  __bf16* Hc = (__bf16*)w; w += bfArr;
  __bf16* Xt = (__bf16*)w; w += bfArr;
  __bf16* Xc = (__bf16*)w; w += bfArr;
  const size_t stArr = (size_t)kB * kHW * sizeof(float);         // 32 KB each
  float* mp  = (float*)w; w += stArr;
  float* rdp = (float*)w; w += stArr;
  float* mc  = (float*)w; w += stArr;
  float* rdc = (float*)w; w += stArr;
  const size_t oArr = (size_t)kB * kC * kHW * sizeof(float);     // 4 MB each
  float* Otp = (float*)w; w += oArr;
  float* Otc = (float*)w; w += oArr;

  conv_prep_kernel<<<dim3(kHW / 64, kB), 256, 0, stream>>>(
      x, fw, fb, gw, gb, hw, hb, Ft, Gt, Hc, Xt, Xc);

  attn_stats_kernel<<<dim3(kHW / (16 * 8), kB, 2), 256, 0, stream>>>(
      Gt, Ft, Xt, mp, rdp, mc, rdc);

  attn_apply_kernel<<<dim3(kHW / (16 * 8), kB, 2), 256, 0, stream>>>(
      Gt, Ft, Hc, Xt, Xc, mp, rdp, mc, rdc, Otp, Otc);

  fuse_kernel<<<dim3(kHW / 32, kB), 256, 0, stream>>>(
      x, Otp, Otc, ow, ob, y);
}
